// CrossAttention_55319178772617
// MI455X (gfx1250) — matrix operand order repaired and checked by reading, not compiled, not run on hardware
//
#include <hip/hip_runtime.h>

#define DIM 768
#define NH 12
#define HD 64
#define NX 1024
#define NY 2048
#define BATCH 4
#define SCALE 0.125f

typedef __attribute__((ext_vector_type(16))) __bf16 v16bf;
typedef __attribute__((ext_vector_type(8)))  float  v8f;

union Frag { v16bf v; uint4 q[2]; };

__device__ inline unsigned short f2bf(float x) {
  union { float f; unsigned u; } c; c.f = x;
  unsigned r = c.u + 0x7FFFu + ((c.u >> 16) & 1u);   // round-to-nearest-even
  return (unsigned short)(r >> 16);
}

// A-operand 16x32 bf16 fragment from row-major LDS tile (row stride ld elems).
// ISA layout: lanes 0-15 row M=lane, K={0..7,16..23}; lanes 16-31 K={8..15,24..31}.
__device__ inline v16bf load_a(const unsigned short* p, int ld) {
  const int lane = threadIdx.x & 31;
  const int row  = lane & 15;
  const int half = lane >> 4;
  const unsigned short* b = p + row * ld + half * 8;
  Frag f;
  f.q[0] = *(const uint4*)(b);        // K = half*8 .. +7
  f.q[1] = *(const uint4*)(b + 16);   // K = 16 + half*8 .. +7
  return f.v;
}

// B-operand 32x16 bf16 fragment. LDS holds "Bt": column n contiguous over k
// (row stride ld). Lanes 0-15: col=lane, K={0..7,16..23}; lanes 16-31: K={8..15,24..31}.
__device__ inline v16bf load_b(const unsigned short* p, int ld) {
  const int lane = threadIdx.x & 31;
  const int col  = lane & 15;
  const int half = lane >> 4;
  const unsigned short* b = p + col * ld + half * 8;
  Frag f;
  f.q[0] = *(const uint4*)(b);
  f.q[1] = *(const uint4*)(b + 16);
  return f.v;
}

// max-reduce across the 16-lane group (xor network). Prefer v_permlane16_b32
// (VALU pipe) over ds_bpermute so the LDS pipe stays free for fragments.
#if defined(__has_builtin) && __has_builtin(__builtin_amdgcn_permlane16)
__device__ inline float plx(float v, unsigned s0, unsigned s1) {
  int i = __float_as_int(v);
  return __int_as_float(__builtin_amdgcn_permlane16(i, i, s0, s1, false, false));
}
__device__ inline float redmax16(float v) {
  v = fmaxf(v, plx(v, 0xFEDCBA98u, 0x76543210u));   // xor 8
  v = fmaxf(v, plx(v, 0x32107654u, 0xBA98FEDCu));   // xor 4
  v = fmaxf(v, plx(v, 0x45670123u, 0xCDEF89ABu));   // xor 3
  v = fmaxf(v, plx(v, 0x67452301u, 0xEFCDAB89u));   // xor 1
  return v;
}
#else
__device__ inline float redmax16(float v) {
  v = fmaxf(v, __shfl_xor(v, 8, 32));
  v = fmaxf(v, __shfl_xor(v, 4, 32));
  v = fmaxf(v, __shfl_xor(v, 2, 32));
  v = fmaxf(v, __shfl_xor(v, 1, 32));
  return v;
}
#endif

// ---------------------------------------------------------------------------
// C[M x 768] = A[M x 768] @ W[768 x 768]^T (+ bias). A is f32 or bf16.
// 128 threads (4 waves), block tile 64x64, double-buffered LDS pipeline.
// ---------------------------------------------------------------------------
template <bool A_F32, bool OUT_F32>
__global__ __launch_bounds__(128)
void gemm_wt(const void* __restrict__ Ap, const float* __restrict__ W,
             const float* __restrict__ bias, void* __restrict__ Cp) {
  __shared__ unsigned short As[2][64 * 40];
  __shared__ unsigned short Ws[2][64 * 40];
  const int tid  = threadIdx.x;
  const int wave = tid >> 5;
  const int lane = tid & 31;
  const int n0   = blockIdx.x * 64;
  const int m0   = blockIdx.y * 64;

  const float*          Af = (const float*)Ap;
  const unsigned short* Ab = (const unsigned short*)Ap;

  v8f acc[4];
#pragma unroll
  for (int j = 0; j < 4; ++j)
#pragma unroll
    for (int e = 0; e < 8; ++e) acc[j][e] = 0.f;

  float          aregF[16];
  unsigned short aregH[16];
  float          wreg[16];

  auto g_load = [&](int k0) {
#pragma unroll
    for (int i = 0; i < 16; ++i) {
      int e = tid + i * 128;
      int r = e >> 5, c = e & 31;
      if (A_F32) aregF[i] = Af[(size_t)(m0 + r) * DIM + k0 + c];
      else       aregH[i] = Ab[(size_t)(m0 + r) * DIM + k0 + c];
      wreg[i] = W[(size_t)(n0 + r) * DIM + k0 + c];
    }
  };
  auto l_store = [&](int buf) {
#pragma unroll
    for (int i = 0; i < 16; ++i) {
      int e = tid + i * 128;
      int r = e >> 5, c = e & 31;
      As[buf][r * 40 + c] = A_F32 ? f2bf(aregF[i]) : aregH[i];
      Ws[buf][r * 40 + c] = f2bf(wreg[i]);
    }
  };

  g_load(0);
  l_store(0);
  __syncthreads();

  const int KSTEPS = DIM / 32;   // 24
  for (int it = 0; it < KSTEPS; ++it) {
    const int cur = it & 1;
    if (it + 1 < KSTEPS) g_load((it + 1) * 32);   // overlap global with WMMA

    v16bf a = load_a(As[cur] + wave * 16 * 40, 40);
    v16bf bfr[4];
#pragma unroll
    for (int j = 0; j < 4; ++j) bfr[j] = load_b(Ws[cur] + j * 16 * 40, 40);
#pragma unroll
    for (int j = 0; j < 4; ++j)
      acc[j] = __builtin_amdgcn_wmma_f32_16x16x32_bf16(
          false, a, false, bfr[j], (short)0, acc[j], false, false);

    if (it + 1 < KSTEPS) l_store(1 - cur);
    __syncthreads();
  }

  const int half = lane >> 4;
  const int col  = lane & 15;
#pragma unroll
  for (int j = 0; j < 4; ++j) {
    const int n = n0 + j * 16 + col;
    const float bv = OUT_F32 ? bias[n] : 0.f;
#pragma unroll
    for (int e = 0; e < 8; ++e) {
      const int m = m0 + wave * 16 + e + half * 8;   // C/D layout row
      if (OUT_F32) ((float*)Cp)[(size_t)m * DIM + n] = acc[j][e] + bv;
      else ((unsigned short*)Cp)[(size_t)m * DIM + n] = f2bf(acc[j][e]);
    }
  }
}

// ---------------------------------------------------------------------------
// Flash attention: grid (NX/64, NH, BATCH), 128 threads (4 waves).
// Wave owns 16 q-rows; streams Ny in 64-key chunks (double-buffered LDS).
// Row sums of P computed with a ones-operand WMMA; row max via permlane.
// ---------------------------------------------------------------------------
__global__ __launch_bounds__(128)
void attn_kernel(const unsigned short* __restrict__ qb,
                 const unsigned short* __restrict__ kb,
                 const unsigned short* __restrict__ vb,
                 unsigned short* __restrict__ ob) {
  __shared__ unsigned short Qs[64 * 72];
  __shared__ unsigned short Ks[2 * 64 * 72];
  __shared__ unsigned short Vt[2 * 64 * 72];       // transposed: Vt[d][key]
  __shared__ unsigned short Pw[4 * 16 * 72];       // per-wave P staging

  const int tid  = threadIdx.x;
  const int wave = tid >> 5;
  const int lane = tid & 31;
  const int half = lane >> 4;
  const int col  = lane & 15;

  const int q0 = blockIdx.x * 64;
  const int h  = blockIdx.y;
  const int b  = blockIdx.z;

  const size_t qbase  = ((size_t)b * NX + q0) * DIM + h * HD;
  const size_t kvbase = ((size_t)b * NY) * DIM + h * HD;

  uint4 kreg[4], vreg[4];
  auto g_load_kv = [&](int kk) {
#pragma unroll
    for (int i = 0; i < 4; ++i) {
      int e = tid + i * 128;
      int r = e >> 3, c8 = e & 7;
      kreg[i] = *(const uint4*)(kb + kvbase + (size_t)(kk + r) * DIM + c8 * 8);
      vreg[i] = *(const uint4*)(vb + kvbase + (size_t)(kk + r) * DIM + c8 * 8);
    }
  };
  auto l_store_kv = [&](int buf) {
#pragma unroll
    for (int i = 0; i < 4; ++i) {
      int e = tid + i * 128;
      int r = e >> 3, c8 = e & 7;
      *(uint4*)(Ks + buf * (64 * 72) + r * 72 + c8 * 8) = kreg[i];
      const unsigned* w = (const unsigned*)&vreg[i];
#pragma unroll
      for (int j = 0; j < 8; ++j) {               // transpose on store
        unsigned short hv = (unsigned short)(w[j >> 1] >> ((j & 1) * 16));
        Vt[buf * (64 * 72) + (c8 * 8 + j) * 72 + r] = hv;
      }
    }
  };

  // Q tile 64x64 copy + first K/V chunk
#pragma unroll
  for (int i = 0; i < 4; ++i) {
    int e = tid + i * 128;
    int r = e >> 3, c8 = e & 7;
    *(uint4*)(Qs + r * 72 + c8 * 8) =
        *(const uint4*)(qb + qbase + (size_t)r * DIM + c8 * 8);
  }
  g_load_kv(0);
  l_store_kv(0);
  __syncthreads();

  const v16bf aQ0 = load_a(Qs + wave * 16 * 72, 72);       // d = 0..31
  const v16bf aQ1 = load_a(Qs + wave * 16 * 72 + 32, 72);  // d = 32..63

  // constant all-ones B operand (bf16 1.0 = 0x3F80) for P row sums
  Frag onesf;
  {
    uint4 o; o.x = o.y = o.z = o.w = 0x3F803F80u;
    onesf.q[0] = o; onesf.q[1] = o;
  }
  const v16bf vones = onesf.v;

  v8f O[4], lacc;
  float mrun[8];
#pragma unroll
  for (int j = 0; j < 4; ++j)
#pragma unroll
    for (int e = 0; e < 8; ++e) O[j][e] = 0.f;
#pragma unroll
  for (int e = 0; e < 8; ++e) { lacc[e] = 0.f; mrun[e] = -1e30f; }

  unsigned short* Pme = Pw + wave * 16 * 72;

  const int CHUNKS = NY / 64;   // 32
  for (int it = 0; it < CHUNKS; ++it) {
    const int cur = it & 1;
    const unsigned short* KsB = Ks + cur * (64 * 72);
    const unsigned short* VtB = Vt + cur * (64 * 72);

    if (it + 1 < CHUNKS) g_load_kv((it + 1) * 64);  // overlap global with math

    // S(16x64) = Q Kchunk^T  (raw; SCALE folded into the exp below)
    v8f S[4];
#pragma unroll
    for (int tp = 0; tp < 2; ++tp) {
      v16bf bfr[4];
#pragma unroll
      for (int tt = 0; tt < 2; ++tt) {
        const int t = tp * 2 + tt;
        bfr[2 * tt]     = load_b(KsB + t * 16 * 72, 72);
        bfr[2 * tt + 1] = load_b(KsB + t * 16 * 72 + 32, 72);
      }
#pragma unroll
      for (int tt = 0; tt < 2; ++tt) {
        const int t = tp * 2 + tt;
        v8f z;
#pragma unroll
        for (int e = 0; e < 8; ++e) z[e] = 0.f;
        z = __builtin_amdgcn_wmma_f32_16x16x32_bf16(false, aQ0, false,
                bfr[2 * tt], (short)0, z, false, false);
        S[t] = __builtin_amdgcn_wmma_f32_16x16x32_bf16(false, aQ1, false,
                bfr[2 * tt + 1], (short)0, z, false, false);
      }
    }

    // fold tiles, then one 16-lane max reduction per row (VALU permlane)
#pragma unroll
    for (int e = 0; e < 8; ++e) {
      float mloc = fmaxf(fmaxf(S[0][e], S[1][e]), fmaxf(S[2][e], S[3][e]));
      const float mnew = fmaxf(mrun[e], redmax16(mloc) * SCALE);
      const float alpha = __expf(mrun[e] - mnew);
      lacc[e] *= alpha;
#pragma unroll
      for (int j = 0; j < 4; ++j) O[j][e] *= alpha;
      mrun[e] = mnew;
    }

    // P = exp(S*SCALE - m)  -> wave-private LDS (D-layout -> A-layout)
#pragma unroll
    for (int t = 0; t < 4; ++t)
#pragma unroll
      for (int e = 0; e < 8; ++e) {
        const float p = __expf(fmaf(S[t][e], SCALE, -mrun[e]));
        Pme[(e + half * 8) * 72 + t * 16 + col] = f2bf(p);
      }

    // A-operand views of P (LDS ops are in-order within a wave)
    const v16bf aP0 = load_a(Pme, 72);
    const v16bf aP1 = load_a(Pme + 32, 72);

    // l += P . ones  (row sums on the matrix pipe, same C/D layout as O)
    lacc = __builtin_amdgcn_wmma_f32_16x16x32_bf16(false, aP0, false, vones,
                                                   (short)0, lacc, false, false);
    lacc = __builtin_amdgcn_wmma_f32_16x16x32_bf16(false, aP1, false, vones,
                                                   (short)0, lacc, false, false);

    // O += P V
#pragma unroll
    for (int jp = 0; jp < 2; ++jp) {
      v16bf bfr[4];
#pragma unroll
      for (int jj = 0; jj < 2; ++jj) {
        const int j = jp * 2 + jj;
        bfr[2 * jj]     = load_b(VtB + j * 16 * 72, 72);
        bfr[2 * jj + 1] = load_b(VtB + j * 16 * 72 + 32, 72);
      }
#pragma unroll
      for (int jj = 0; jj < 2; ++jj) {
        const int j = jp * 2 + jj;
        O[j] = __builtin_amdgcn_wmma_f32_16x16x32_bf16(false, aP0, false,
                   bfr[2 * jj], (short)0, O[j], false, false);
        O[j] = __builtin_amdgcn_wmma_f32_16x16x32_bf16(false, aP1, false,
                   bfr[2 * jj + 1], (short)0, O[j], false, false);
      }
    }

    if (it + 1 < CHUNKS) l_store_kv(1 - cur);
    __syncthreads();
  }

#pragma unroll
  for (int e = 0; e < 8; ++e) {
    const float inv = 1.f / lacc[e];
    const int m = q0 + wave * 16 + e + half * 8;
#pragma unroll
    for (int j = 0; j < 4; ++j)
      ob[((size_t)b * NX + m) * DIM + h * HD + j * 16 + col] =
          f2bf(O[j][e] * inv);
  }
}

extern "C" void kernel_launch(void* const* d_in, const int* in_sizes, int n_in,
                              void* d_out, int out_size, void* d_ws, size_t ws_size,
                              hipStream_t stream) {
  (void)in_sizes; (void)n_in; (void)out_size; (void)ws_size;
  const float* x  = (const float*)d_in[0];
  const float* y  = (const float*)d_in[1];
  const float* Wq = (const float*)d_in[2];
  const float* Wk = (const float*)d_in[3];
  const float* Wv = (const float*)d_in[4];
  const float* Wp = (const float*)d_in[5];
  const float* bp = (const float*)d_in[6];

  unsigned short* qb = (unsigned short*)d_ws;                 // [B*NX, DIM] bf16
  unsigned short* kb = qb + (size_t)BATCH * NX * DIM;         // [B*NY, DIM] bf16
  unsigned short* vb = kb + (size_t)BATCH * NY * DIM;         // [B*NY, DIM] bf16
  unsigned short* ab = vb + (size_t)BATCH * NY * DIM;         // [B*NX, DIM] bf16

  dim3 blk(128);
  gemm_wt<true, false><<<dim3(DIM / 64, (BATCH * NX) / 64), blk, 0, stream>>>(x, Wq, nullptr, qb);
  gemm_wt<true, false><<<dim3(DIM / 64, (BATCH * NY) / 64), blk, 0, stream>>>(y, Wk, nullptr, kb);
  gemm_wt<true, false><<<dim3(DIM / 64, (BATCH * NY) / 64), blk, 0, stream>>>(y, Wv, nullptr, vb);
  attn_kernel<<<dim3(NX / 64, NH, BATCH), blk, 0, stream>>>(qb, kb, vb, ab);
  gemm_wt<false, true><<<dim3(DIM / 64, (BATCH * NX) / 64), blk, 0, stream>>>(ab, Wp, bp, (float*)d_out);
}
